// GraphEmbeddingActor_18476949307701
// MI455X (gfx1250) — matrix operand order repaired and checked by reading, not compiled, not run on hardware
//
#include <hip/hip_runtime.h>
#include <hip/hip_fp16.h>

#define N_NODES 100000
#define N_EDGES 1600000
#define D 64

typedef __attribute__((ext_vector_type(16))) _Float16 v16h;
typedef __attribute__((ext_vector_type(8)))  float    v8f;

// ---------------- utility kernels ----------------

__global__ void fill_zero4(float4* __restrict__ p, int n4) {
    int i = blockIdx.x * blockDim.x + threadIdx.x;
    if (i < n4) p[i] = make_float4(0.f, 0.f, 0.f, 0.f);
}

__global__ void degree_kernel(const int* __restrict__ src, const int* __restrict__ dst,
                              float* __restrict__ deg_out, float* __restrict__ deg_in, int e_cnt) {
    int e = blockIdx.x * blockDim.x + threadIdx.x;
    if (e < e_cnt) {
        unsafeAtomicAdd(&deg_out[src[e]], 1.0f);
        unsafeAtomicAdd(&deg_in[dst[e]], 1.0f);
    }
}

__global__ void norm_kernel(float* __restrict__ p, int n) {
    int i = blockIdx.x * blockDim.x + threadIdx.x;
    if (i < n) p[i] = rsqrtf(fmaxf(p[i], 1.0f));
}

// Convert three 64x64 f32 weight matrices to f16, stored TRANSPOSED: wt[l][n*64+k] = W_l[k][n]
__global__ void wprep_kernel(const float* __restrict__ W0, const float* __restrict__ W1,
                             const float* __restrict__ W2, _Float16* __restrict__ wt) {
    int idx = blockIdx.x * blockDim.x + threadIdx.x;
    if (idx >= 3 * 4096) return;
    int l = idx >> 12;
    int w = idx & 4095;
    int n = w >> 6, k = w & 63;
    const float* W = (l == 0) ? W0 : ((l == 1) ? W1 : W2);
    wt[idx] = (_Float16)W[k * 64 + n];
}

// m[dst[e]] += h[src[e]] * norm_src[src[e]]  (16 lanes per edge, 4 cols per lane)
__global__ void aggregate_kernel(const float* __restrict__ h, const int* __restrict__ src,
                                 const int* __restrict__ dst, const float* __restrict__ norm_src,
                                 float* __restrict__ m, int e_cnt) {
    int idx = blockIdx.x * blockDim.x + threadIdx.x;
    int e = idx >> 4;
    if (e >= e_cnt) return;
    int c = (idx & 15) << 2;
    int s = src[e], d = dst[e];
    float ns = norm_src[s];
    const float4 v = *reinterpret_cast<const float4*>(h + (size_t)s * D + c);
    float* mp = m + (size_t)d * D + c;
    unsafeAtomicAdd(mp + 0, v.x * ns);
    unsafeAtomicAdd(mp + 1, v.y * ns);
    unsafeAtomicAdd(mp + 2, v.z * ns);
    unsafeAtomicAdd(mp + 3, v.w * ns);
}

// ---------------- WMMA GEMM + bias + activation ----------------
// out[16-row tile] = act( (m * norm_dst[:,None]) @ W + b )
// One wave per 16-row tile; K=64 via two v_wmma_f32_16x16x32_f16; 4 column tiles.
// ACT: 0 = relu, 1 = sigmoid + 1e-8
template <int ACT>
__global__ void gemm_act_kernel(const float* __restrict__ m, const float* __restrict__ norm_dst,
                                const _Float16* __restrict__ wt, const float* __restrict__ bias,
                                float* __restrict__ out, int ntiles) {
    __shared__ __align__(16) _Float16 wlds[64 * 64];  // Wt[n][k], f16, 8 KB
    {
        const uint4* s4 = reinterpret_cast<const uint4*>(wt);
        uint4* d4 = reinterpret_cast<uint4*>(wlds);
        for (int i = threadIdx.x; i < 512; i += blockDim.x) d4[i] = s4[i];
    }
    __syncthreads();

    int wave = threadIdx.x >> 5;
    int tile = blockIdx.x * 4 + wave;
    if (tile >= ntiles) return;

    int lane = threadIdx.x & 31;
    int hsel = lane >> 4;   // 0: lanes 0-15, 1: lanes 16-31
    int r    = lane & 15;
    int r0   = tile << 4;
    int row  = r0 + r;

    float nd = norm_dst[row];
    const float* mrow = m + (size_t)row * D;

    // A fragments: lane holds K = hsel*16 + e (contiguous, matching B). a1 is K-tile +32.
    v16h a0, a1;
#pragma unroll
    for (int e = 0; e < 16; ++e) {
        int koff = e + hsel * 16;
        a0[e] = (_Float16)(mrow[koff] * nd);
        a1[e] = (_Float16)(mrow[32 + koff] * nd);
    }

#pragma unroll
    for (int nt = 0; nt < 4; ++nt) {
        int n = nt * 16 + r;  // output column for this lane
        // B fragments: lane holds column n, K = ktile*32 + hsel*16 + e  (contiguous in Wt[n][k])
        const _Float16* wp0 = &wlds[n * 64 + hsel * 16];
        const _Float16* wp1 = wp0 + 32;
        v16h b0, b1;
#pragma unroll
        for (int e = 0; e < 16; ++e) { b0[e] = wp0[e]; b1[e] = wp1[e]; }

        v8f c = {};
        c = __builtin_amdgcn_wmma_f32_16x16x32_f16(false, a0, false, b0, (short)0, c, false, false);
        c = __builtin_amdgcn_wmma_f32_16x16x32_f16(false, a1, false, b1, (short)0, c, false, false);

        float bn = bias[n];
#pragma unroll
        for (int i = 0; i < 8; ++i) {
            int orow = r0 + hsel * 8 + i;  // C/D layout: VGPR i -> M=i (lanes<16) / M=8+i
            float v = c[i] + bn;
            if (ACT == 0) v = fmaxf(v, 0.0f);
            else          v = 1.0f / (1.0f + __expf(-v)) + 1e-8f;
            out[(size_t)orow * D + n] = v;
        }
    }
}

// ---------------- launcher ----------------

extern "C" void kernel_launch(void* const* d_in, const int* in_sizes, int n_in,
                              void* d_out, int out_size, void* d_ws, size_t ws_size,
                              hipStream_t stream) {
    const float* features = (const float*)d_in[0];
    const int*   src      = (const int*)d_in[1];
    const int*   dst      = (const int*)d_in[2];
    const float* W0 = (const float*)d_in[3];
    const float* b0 = (const float*)d_in[4];
    const float* W1 = (const float*)d_in[5];
    const float* b1 = (const float*)d_in[6];
    const float* W2 = (const float*)d_in[7];
    const float* b2 = (const float*)d_in[8];
    float* out = (float*)d_out;

    float* ws        = (float*)d_ws;
    float* norm_src  = ws;                               // N
    float* norm_dst  = ws + N_NODES;                     // N
    float* m         = ws + 2 * (size_t)N_NODES;         // 64*N
    float* hA        = ws + 66 * (size_t)N_NODES;        // 64*N
    _Float16* wt     = (_Float16*)(ws + 130 * (size_t)N_NODES);  // 3*4096 halves

    const int ntiles = N_NODES / 16;                 // 6250 (exact)
    const int gemm_blocks = (ntiles + 3) / 4;        // 4 waves/block
    const int m4 = (D * N_NODES) / 4;                // float4 count for m

    // degrees -> norms (in place)
    fill_zero4<<<((2 * N_NODES / 4) + 255) / 256, 256, 0, stream>>>((float4*)norm_src, 2 * N_NODES / 4);
    degree_kernel<<<(N_EDGES + 255) / 256, 256, 0, stream>>>(src, dst, norm_src, norm_dst, N_EDGES);
    norm_kernel<<<(2 * N_NODES + 255) / 256, 256, 0, stream>>>(norm_src, 2 * N_NODES);

    // weights -> f16 transposed
    wprep_kernel<<<(3 * 4096 + 255) / 256, 256, 0, stream>>>(W0, W1, W2, wt);

    const int agg_blocks = ((N_EDGES * 16) + 255) / 256;

    // layer 0: features -> hA (relu)
    fill_zero4<<<(m4 + 255) / 256, 256, 0, stream>>>((float4*)m, m4);
    aggregate_kernel<<<agg_blocks, 256, 0, stream>>>(features, src, dst, norm_src, m, N_EDGES);
    gemm_act_kernel<0><<<gemm_blocks, 128, 0, stream>>>(m, norm_dst, wt, b0, hA, ntiles);

    // layer 1: hA -> hA (relu)  (aggregate fully consumes hA before gemm rewrites it)
    fill_zero4<<<(m4 + 255) / 256, 256, 0, stream>>>((float4*)m, m4);
    aggregate_kernel<<<agg_blocks, 256, 0, stream>>>(hA, src, dst, norm_src, m, N_EDGES);
    gemm_act_kernel<0><<<gemm_blocks, 128, 0, stream>>>(m, norm_dst, wt + 4096, b1, hA, ntiles);

    // layer 2: hA -> out (sigmoid + 1e-8)
    fill_zero4<<<(m4 + 255) / 256, 256, 0, stream>>>((float4*)m, m4);
    aggregate_kernel<<<agg_blocks, 256, 0, stream>>>(hA, src, dst, norm_src, m, N_EDGES);
    gemm_act_kernel<1><<<gemm_blocks, 128, 0, stream>>>(m, norm_dst, wt + 2 * 4096, b2, out, ntiles);
}
